// FC_39006892982711
// MI455X (gfx1250) — matrix operand order repaired and checked by reading, not compiled, not run on hardware
//
#include <hip/hip_runtime.h>
#include <hip/hip_bf16.h>

// ---------------------------------------------------------------------------
// Types for CDNA5 WMMA (gfx1250, wave32)
// ---------------------------------------------------------------------------
typedef __attribute__((ext_vector_type(16))) __bf16 v16bf;
typedef __attribute__((ext_vector_type(8)))  float  v8f;

#define EPSBN       1e-5f
#define LEAKY_SLOPE 0.01f
#define RRELU_SLOPE ((1.0f/8.0f + 1.0f/3.0f) * 0.5f)

__device__ __forceinline__ unsigned short f2bf(float f) {
  unsigned int u = __float_as_uint(f);
  unsigned int r = (u + 0x7FFFu + ((u >> 16) & 1u)) >> 16;   // RNE
  return (unsigned short)r;
}
__device__ __forceinline__ float bf2f(unsigned short h) {
  return __uint_as_float(((unsigned int)h) << 16);
}

// ---------------------------------------------------------------------------
// Utility kernels
// ---------------------------------------------------------------------------
__global__ void fill_f32(float* p, float v, long n) {
  long i = (long)blockIdx.x * blockDim.x + threadIdx.x;
  if (i < n) p[i] = v;
}
__global__ void fill_i32(int* p, int v, long n) {
  long i = (long)blockIdx.x * blockDim.x + threadIdx.x;
  if (i < n) p[i] = v;
}
__global__ void cvt_f32_bf16(const float* __restrict__ in,
                             unsigned short* __restrict__ out, long n) {
  long i = (long)blockIdx.x * blockDim.x + threadIdx.x;
  if (i < n) out[i] = f2bf(in[i]);
}

// deg[dst] += w  (deg pre-initialized to 2.0 = improved self-loop weight)
__global__ void deg_accum(const int* __restrict__ dst,
                          const float* __restrict__ w, long E,
                          float* __restrict__ deg) {
  long e = (long)blockIdx.x * blockDim.x + threadIdx.x;
  if (e < E) atomicAdd(&deg[dst[e]], w[e]);
}
__global__ void deg_to_dinv(float* deg, long n) {
  long i = (long)blockIdx.x * blockDim.x + threadIdx.x;
  if (i < n) { float d = deg[i]; deg[i] = d > 0.0f ? rsqrtf(d) : 0.0f; }
}

// Assign compact slots to gathered nodes (slot pre-filled with -1).
__global__ void mark_nodes(const int* __restrict__ idx, int B,
                           int* slot, int* counter) {
  int b = blockIdx.x * blockDim.x + threadIdx.x;
  if (b >= B) return;
  int node = idx[b];
  if (atomicCAS(&slot[node], -1, -2) == -1)
    slot[node] = atomicAdd(counter, 1);
}

// Filtered GCN aggregation, one edge per wave (8 edges per 256-thread block).
__global__ __launch_bounds__(256)
void gcn_aggregate(const int* __restrict__ src, const int* __restrict__ dst,
                   const float* __restrict__ w, long E,
                   const float* __restrict__ dinv, const int* __restrict__ slot,
                   const unsigned short* __restrict__ h, int F,
                   float* __restrict__ outc) {
  const int wid  = threadIdx.x >> 5;
  const int lane = threadIdx.x & 31;
  long e = (long)blockIdx.x * 8 + wid;
  if (e >= E) return;
  int d  = dst[e];
  int id = slot[d];
  if (id < 0) return;                 // dst not gathered -> row never read
  int s = src[e];
  float norm = dinv[s] * w[e] * dinv[d];
  const unsigned short* hrow = h + (size_t)s * F;
  float* orow = outc + (size_t)id * F;
  for (int c0 = lane * 4; c0 < F; c0 += 128) {     // 4 bf16 per lane per step
    uint2 v = *(const uint2*)(hrow + c0);
    atomicAdd(&orow[c0 + 0], bf2f((unsigned short)(v.x & 0xFFFFu)) * norm);
    atomicAdd(&orow[c0 + 1], bf2f((unsigned short)(v.x >> 16)) * norm);
    atomicAdd(&orow[c0 + 2], bf2f((unsigned short)(v.y & 0xFFFFu)) * norm);
    atomicAdd(&orow[c0 + 3], bf2f((unsigned short)(v.y >> 16)) * norm);
  }
}

// Self-loop (weight 2.0) + bias + leaky + gather into bf16 feat slice.
__global__ __launch_bounds__(256)
void gcn_finish_gather(const int* __restrict__ idx, const int* __restrict__ slot,
                       const float* __restrict__ dinv,
                       const unsigned short* __restrict__ h,
                       const float* __restrict__ outc,
                       const float* __restrict__ bias, int F,
                       unsigned short* __restrict__ feat, int stride, int colOff) {
  int b = blockIdx.x;
  int node = idx[b];
  int id = slot[node];
  float dn = dinv[node];
  float selfn = 2.0f * dn * dn;
  const unsigned short* hrow = h + (size_t)node * F;
  const float* orow = outc + (size_t)id * F;
  unsigned short* frow = feat + (size_t)b * stride + colOff;
  for (int c = threadIdx.x; c < F; c += blockDim.x) {
    float v = orow[c] + selfn * bf2f(hrow[c]) + bias[c];
    v = v > 0.0f ? v : LEAKY_SLOPE * v;
    frow[c] = f2bf(v);
  }
}

// f32 [B,cols] -> bf16 slice of feat
__global__ __launch_bounds__(256)
void copy_feat(const float* __restrict__ in, int cols,
               unsigned short* __restrict__ feat, int stride, int colOff) {
  int b = blockIdx.x;
  const float* irow = in + (size_t)b * cols;
  unsigned short* frow = feat + (size_t)b * stride + colOff;
  for (int c = threadIdx.x; c < cols; c += blockDim.x)
    frow[c] = f2bf(irow[c]);
}

// y[b] = dot(o[b,0:128], W4) + b4   (full f32)
__global__ void final_fc(const float* __restrict__ o, const float* __restrict__ W4,
                         const float* __restrict__ b4, float* __restrict__ y, int B) {
  int b = blockIdx.x * blockDim.x + threadIdx.x;
  if (b >= B) return;
  const float* row = o + (size_t)b * 128;
  float s = b4[0];
  #pragma unroll 8
  for (int c = 0; c < 128; ++c) s += row[c] * W4[c];
  y[b] = s;
}

// ---------------------------------------------------------------------------
// 128x128x32 double-buffered bf16 WMMA GEMM: C = A[MxK] @ B[KxN] (row-major)
// 8 waves/block (wave32), each wave owns a 64x32 tile = 4x2 WMMA frags.
// Branchless staging: clamped addresses + select-zero (no exec divergence).
// MODE 0: store bf16 C                          (GCN h)
// MODE 1: +bias, BN, leaky -> f32 and/or bf16   (L1, L2)
// MODE 2: +bias, leaky(slope), BN -> f32        (L3, RReLU eval)
// ---------------------------------------------------------------------------
union FragBF { int i[8]; v16bf v; };

template <int MODE>
__global__ __launch_bounds__(256)
void wmma_gemm_bf16(const unsigned short* __restrict__ A, int lda,
                    const unsigned short* __restrict__ Bm, int ldb,
                    int M, int N, int K,
                    float* __restrict__ Cf, int ldcf,
                    unsigned short* __restrict__ Cb, int ldcb,
                    const float* __restrict__ bias,
                    const float* __restrict__ gamma, const float* __restrict__ beta,
                    const float* __restrict__ mean,  const float* __restrict__ var,
                    float slope) {
  // stride 36 halfwords (72B): b64-store aligned, 18*r mod 64 injective (r<16)
  __shared__ __align__(16) unsigned short As[2][128][36];   // [m][k]
  __shared__ __align__(16) unsigned short Bs[2][128][36];   // transposed [n][k]

  const int tid   = threadIdx.x;
  const int lane  = tid & 31;
  const int wid   = tid >> 5;
  const int wm    = wid >> 2;          // 0..1 : 64-row strip
  const int wn    = wid & 3;           // 0..3 : 32-col strip
  const int row16 = lane & 15;
  const int khalf = lane >> 4;

  const int bm = blockIdx.y * 128;
  const int bn = blockIdx.x * 128;

  v8f acc[4][2] = {};

  const int ktiles = (K + 31) / 32;

  // Branchless staging of one 128x32 A tile + 32x128 B tile into buffer `buf`.
  auto stage = [&](int buf, int k0) {
    // ---- A: 4 x uint2 (4 bf16) per thread, unconditional clamped loads ----
    uint2 av[4]; int arow[4], acq[4];
    #pragma unroll
    for (int i = 0; i < 4; ++i) {
      int p  = i * 256 + tid;
      int r  = p >> 3;                 // 0..127
      int cq = p & 7;                  // chunk of 4 bf16 (K % 4 == 0 always)
      int gr = bm + r;
      int gk = k0 + cq * 4;
      bool ok = (gr < M) && (gk < K);
      int grc = gr < M ? gr : M - 1;
      int gkc = gk < K ? gk : 0;
      uint2 v = *(const uint2*)(A + (size_t)grc * lda + gkc);
      if (i == 0 && k0 + 64 < K) __builtin_prefetch(A + (size_t)grc * lda + gk + 64, 0, 3);
      if (!ok) { v.x = 0u; v.y = 0u; }
      av[i] = v; arow[i] = r; acq[i] = cq;
    }
    // ---- B: 2 x uint4 (8 bf16) per thread ----
    uint4 bv[2]; int bkr[2], bn0[2];
    #pragma unroll
    for (int i = 0; i < 2; ++i) {
      int p  = i * 256 + tid;
      int kr = p >> 4;                 // 0..31
      int cq = p & 15;                 // chunk of 8 bf16 (N % 128 == 0 always)
      int gk = k0 + kr;
      bool ok = gk < K;
      int gkc = ok ? gk : 0;
      uint4 v = *(const uint4*)(Bm + (size_t)gkc * ldb + bn + cq * 8);
      if (!ok) { v.x = 0u; v.y = 0u; v.z = 0u; v.w = 0u; }
      bv[i] = v; bkr[i] = kr; bn0[i] = cq * 8;
    }
    // ---- LDS stores ----
    #pragma unroll
    for (int i = 0; i < 4; ++i)
      *(uint2*)(&As[buf][arow[i]][acq[i] * 4]) = av[i];
    #pragma unroll
    for (int i = 0; i < 2; ++i) {
      int kr = bkr[i], n0 = bn0[i];
      Bs[buf][n0 + 0][kr] = (unsigned short)(bv[i].x & 0xFFFFu);
      Bs[buf][n0 + 1][kr] = (unsigned short)(bv[i].x >> 16);
      Bs[buf][n0 + 2][kr] = (unsigned short)(bv[i].y & 0xFFFFu);
      Bs[buf][n0 + 3][kr] = (unsigned short)(bv[i].y >> 16);
      Bs[buf][n0 + 4][kr] = (unsigned short)(bv[i].z & 0xFFFFu);
      Bs[buf][n0 + 5][kr] = (unsigned short)(bv[i].z >> 16);
      Bs[buf][n0 + 6][kr] = (unsigned short)(bv[i].w & 0xFFFFu);
      Bs[buf][n0 + 7][kr] = (unsigned short)(bv[i].w >> 16);
    }
  };

  stage(0, 0);
  for (int kt = 0; kt < ktiles; ++kt) {
    __syncthreads();                       // tile kt ready in buffer kt&1
    if (kt + 1 < ktiles) stage((kt + 1) & 1, (kt + 1) * 32);
    const int buf = kt & 1;

    // ---- fragments per ISA 7.12.2 bf16 layouts ----
    FragBF afr[4], bfr[2];
    #pragma unroll
    for (int mt = 0; mt < 4; ++mt) {
      const unsigned short* ar = &As[buf][wm * 64 + mt * 16 + row16][0];
      #pragma unroll
      for (int v = 0; v < 4; ++v) {
        afr[mt].i[v]     = *(const int*)(ar + khalf * 8 + 2 * v);        // K 0..7 / 8..15
        afr[mt].i[4 + v] = *(const int*)(ar + 16 + khalf * 8 + 2 * v);   // K 16..23 / 24..31
      }
    }
    #pragma unroll
    for (int nt = 0; nt < 2; ++nt) {
      const unsigned short* br = &Bs[buf][wn * 32 + nt * 16 + row16][0];
      #pragma unroll
      for (int v = 0; v < 8; ++v)
        bfr[nt].i[v] = *(const int*)(br + (v >> 2) * 16 + khalf * 8 + 2 * (v & 3));   // K 0..7,16..23 / 8..15,24..31
    }
    // ---- 8 WMMAs per K tile per wave ----
    #pragma unroll
    for (int mt = 0; mt < 4; ++mt)
      #pragma unroll
      for (int nt = 0; nt < 2; ++nt)
        acc[mt][nt] = __builtin_amdgcn_wmma_f32_16x16x32_bf16(
            false, afr[mt].v, false, bfr[nt].v, (short)0, acc[mt][nt],
            false, false);
  }

  // ---- epilogue: C layout = lane&15 -> N, vgpr r -> M = r + 8*khalf ----
  #pragma unroll
  for (int mt = 0; mt < 4; ++mt) {
    #pragma unroll
    for (int nt = 0; nt < 2; ++nt) {
      int gn = bn + wn * 32 + nt * 16 + row16;
      #pragma unroll
      for (int r = 0; r < 8; ++r) {
        int gm = bm + wm * 64 + mt * 16 + khalf * 8 + r;
        if (gm >= M) continue;
        float x = acc[mt][nt][r];
        if (MODE == 0) {
          Cb[(size_t)gm * ldcb + gn] = f2bf(x);
        } else if (MODE == 1) {
          x += bias[gn];
          x = (x - mean[gn]) * rsqrtf(var[gn] + EPSBN) * gamma[gn] + beta[gn];
          x = x > 0.0f ? x : LEAKY_SLOPE * x;
          if (Cf) Cf[(size_t)gm * ldcf + gn] = x;
          if (Cb) Cb[(size_t)gm * ldcb + gn] = f2bf(x);
        } else {  // MODE 2: RReLU(eval) then BN
          x += bias[gn];
          x = x > 0.0f ? x : slope * x;
          x = (x - mean[gn]) * rsqrtf(var[gn] + EPSBN) * gamma[gn] + beta[gn];
          Cf[(size_t)gm * ldcf + gn] = x;
        }
      }
    }
  }
}

// ---------------------------------------------------------------------------
// Host-side orchestration
// ---------------------------------------------------------------------------
extern "C" void kernel_launch(void* const* d_in, const int* in_sizes, int n_in,
                              void* d_out, int out_size, void* d_ws, size_t ws_size,
                              hipStream_t stream) {
  (void)n_in; (void)out_size; (void)ws_size;
  const int*   d_index = (const int*)  d_in[0];
  const int*   p_index = (const int*)  d_in[1];
  const float* d_vecs  = (const float*)d_in[2];
  const float* p_emb   = (const float*)d_in[3];
  const float* d_ecfps = (const float*)d_in[4];
  const int*   d_ei    = (const int*)  d_in[5];
  const float* d_ew    = (const float*)d_in[6];
  const float* p_gos   = (const float*)d_in[7];
  const int*   p_ei    = (const int*)  d_in[8];
  const float* p_ew    = (const float*)d_in[9];
  const float* Wd = (const float*)d_in[10]; const float* bd = (const float*)d_in[11];
  const float* Wp = (const float*)d_in[12]; const float* bp = (const float*)d_in[13];
  const float* W1 = (const float*)d_in[14]; const float* b1 = (const float*)d_in[15];
  const float* g1 = (const float*)d_in[16]; const float* be1= (const float*)d_in[17];
  const float* m1 = (const float*)d_in[18]; const float* v1 = (const float*)d_in[19];
  const float* W2 = (const float*)d_in[20]; const float* b2 = (const float*)d_in[21];
  const float* g2 = (const float*)d_in[22]; const float* be2= (const float*)d_in[23];
  const float* m2 = (const float*)d_in[24]; const float* v2 = (const float*)d_in[25];
  const float* W3 = (const float*)d_in[26]; const float* b3 = (const float*)d_in[27];
  const float* g3 = (const float*)d_in[28]; const float* be3= (const float*)d_in[29];
  const float* m3 = (const float*)d_in[30]; const float* v3 = (const float*)d_in[31];
  const float* W4 = (const float*)d_in[32]; const float* b4 = (const float*)d_in[33];

  const int  B    = in_sizes[0];
  const int  ND   = in_sizes[4] / 1024;
  const int  NPn  = in_sizes[7] / 2812;
  const long ED   = in_sizes[5] / 2;
  const long EP   = in_sizes[8] / 2;
  const int  FD   = 1024, FPK = 2812, FO = 1024;
  const int  FEAT = 3372, H1 = 1024, H2 = 512, H3 = 128;

  // ---- carve workspace ----
  char* wp = (char*)d_ws;
  auto take = [&](size_t bytes) -> void* {
    void* q = (void*)wp;
    wp += (bytes + 255) & ~(size_t)255;
    return q;
  };
  unsigned short* Xd_b = (unsigned short*)take((size_t)ND * FD  * 2);
  unsigned short* Xp_b = (unsigned short*)take((size_t)NPn * FPK * 2);
  unsigned short* Wd_b = (unsigned short*)take((size_t)FD  * FO * 2);
  unsigned short* Wp_b = (unsigned short*)take((size_t)FPK * FO * 2);
  unsigned short* W1_b = (unsigned short*)take((size_t)FEAT* H1 * 2);
  unsigned short* W2_b = (unsigned short*)take((size_t)H1  * H2 * 2);
  unsigned short* W3_b = (unsigned short*)take((size_t)H2  * H3 * 2);
  unsigned short* h_d  = (unsigned short*)take((size_t)ND  * FO * 2);
  unsigned short* h_p  = (unsigned short*)take((size_t)NPn * FO * 2);
  float* dinv_d = (float*)take((size_t)ND  * 4);
  float* dinv_p = (float*)take((size_t)NPn * 4);
  int*   slot_d = (int*)  take((size_t)ND  * 4);
  int*   slot_p = (int*)  take((size_t)NPn * 4);
  int*   ctrs   = (int*)  take(2 * 4);
  float* outc_d = (float*)take((size_t)B * FO * 4);
  float* outc_p = (float*)take((size_t)B * FO * 4);
  unsigned short* feat = (unsigned short*)take((size_t)B * FEAT * 2);
  unsigned short* h1   = (unsigned short*)take((size_t)B * H1 * 2);
  unsigned short* h2   = (unsigned short*)take((size_t)B * H2 * 2);
  float* o3 = (float*)take((size_t)B * H3 * 4);

  float* y_out    = (float*)d_out;       // [B]
  float* feat_out = (float*)d_out + B;   // [B, 512]

  const int T = 256;
  auto nb = [](long n, int t) { return (unsigned)((n + t - 1) / t); };

  // ---- 1. init scratch state (every call) ----
  fill_f32<<<nb(ND, T), T, 0, stream>>>(dinv_d, 2.0f, ND);
  fill_f32<<<nb(NPn, T), T, 0, stream>>>(dinv_p, 2.0f, NPn);
  fill_i32<<<nb(ND, T), T, 0, stream>>>(slot_d, -1, ND);
  fill_i32<<<nb(NPn, T), T, 0, stream>>>(slot_p, -1, NPn);
  fill_i32<<<1, 32, 0, stream>>>(ctrs, 0, 2);
  fill_f32<<<nb((long)B * FO, T), T, 0, stream>>>(outc_d, 0.0f, (long)B * FO);
  fill_f32<<<nb((long)B * FO, T), T, 0, stream>>>(outc_p, 0.0f, (long)B * FO);

  // ---- 2. degrees -> dinv ----
  deg_accum<<<nb(ED, T), T, 0, stream>>>(d_ei + ED, d_ew, ED, dinv_d);
  deg_accum<<<nb(EP, T), T, 0, stream>>>(p_ei + EP, p_ew, EP, dinv_p);
  deg_to_dinv<<<nb(ND, T), T, 0, stream>>>(dinv_d, ND);
  deg_to_dinv<<<nb(NPn, T), T, 0, stream>>>(dinv_p, NPn);

  // ---- 3. mark gathered nodes (compact slots) ----
  mark_nodes<<<nb(B, T), T, 0, stream>>>(d_index, B, slot_d, ctrs + 0);
  mark_nodes<<<nb(B, T), T, 0, stream>>>(p_index, B, slot_p, ctrs + 1);

  // ---- 4. f32 -> bf16 conversions ----
  cvt_f32_bf16<<<nb((long)ND * FD, T), T, 0, stream>>>(d_ecfps, Xd_b, (long)ND * FD);
  cvt_f32_bf16<<<nb((long)NPn * FPK, T), T, 0, stream>>>(p_gos, Xp_b, (long)NPn * FPK);
  cvt_f32_bf16<<<nb((long)FD * FO, T), T, 0, stream>>>(Wd, Wd_b, (long)FD * FO);
  cvt_f32_bf16<<<nb((long)FPK * FO, T), T, 0, stream>>>(Wp, Wp_b, (long)FPK * FO);
  cvt_f32_bf16<<<nb((long)FEAT * H1, T), T, 0, stream>>>(W1, W1_b, (long)FEAT * H1);
  cvt_f32_bf16<<<nb((long)H1 * H2, T), T, 0, stream>>>(W2, W2_b, (long)H1 * H2);
  cvt_f32_bf16<<<nb((long)H2 * H3, T), T, 0, stream>>>(W3, W3_b, (long)H2 * H3);

  // ---- 5. GCN feature transforms: h = X @ W  (WMMA bf16) ----
  {
    dim3 g(FO / 128, (ND + 127) / 128);
    wmma_gemm_bf16<0><<<g, T, 0, stream>>>(Xd_b, FD, Wd_b, FO, ND, FO, FD,
        nullptr, 0, h_d, FO, nullptr, nullptr, nullptr, nullptr, nullptr, 0.f);
  }
  {
    dim3 g(FO / 128, (NPn + 127) / 128);
    wmma_gemm_bf16<0><<<g, T, 0, stream>>>(Xp_b, FPK, Wp_b, FO, NPn, FO, FPK,
        nullptr, 0, h_p, FO, nullptr, nullptr, nullptr, nullptr, nullptr, 0.f);
  }

  // ---- 6. filtered edge aggregation into compact rows (1 edge per wave) ----
  gcn_aggregate<<<nb(ED, 8), T, 0, stream>>>(d_ei, d_ei + ED, d_ew, ED,
                                             dinv_d, slot_d, h_d, FO, outc_d);
  gcn_aggregate<<<nb(EP, 8), T, 0, stream>>>(p_ei, p_ei + EP, p_ew, EP,
                                             dinv_p, slot_p, h_p, FO, outc_p);

  // ---- 7. self-loop + bias + leaky + gather -> feat slices (bf16) ----
  gcn_finish_gather<<<B, T, 0, stream>>>(d_index, slot_d, dinv_d, h_d, outc_d,
                                         bd, FO, feat, FEAT, 300 + 1024);
  gcn_finish_gather<<<B, T, 0, stream>>>(p_index, slot_p, dinv_p, h_p, outc_p,
                                         bp, FO, feat, FEAT, 300 + 1024 + 1024);
  copy_feat<<<B, T, 0, stream>>>(d_vecs, 300, feat, FEAT, 0);
  copy_feat<<<B, T, 0, stream>>>(p_emb, 1024, feat, FEAT, 300);

  // ---- 8. MLP ----
  {  // L1: feat @ W1 -> BN+leaky -> h1 (bf16)
    dim3 g(H1 / 128, (B + 127) / 128);
    wmma_gemm_bf16<1><<<g, T, 0, stream>>>(feat, FEAT, W1_b, H1, B, H1, FEAT,
        nullptr, 0, h1, H1, b1, g1, be1, m1, v1, 0.f);
  }
  {  // L2: h1 @ W2 -> BN+leaky -> feature (f32 to d_out) + h2 (bf16)
    dim3 g(H2 / 128, (B + 127) / 128);
    wmma_gemm_bf16<1><<<g, T, 0, stream>>>(h1, H1, W2_b, H2, B, H2, H1,
        feat_out, H2, h2, H2, b2, g2, be2, m2, v2, 0.f);
  }
  {  // L3: h2 @ W3 -> RReLU(eval) -> BN -> o3 (f32)
    dim3 g(H3 / 128, (B + 127) / 128);
    wmma_gemm_bf16<2><<<g, T, 0, stream>>>(h2, H2, W3_b, H3, B, H3, H2,
        o3, H3, nullptr, 0, b3, g3, be3, m3, v3, RRELU_SLOPE);
  }
  // ---- 9. y = o3 @ W4 + b4 (f32) ----
  final_fc<<<nb(B, T), T, 0, stream>>>(o3, W4, b4, y_out, B);
}
